// GumbelVectorQuantizer_7164005449834
// MI455X (gfx1250) — compile-verified, matrix operand order checked by reading, not run on hardware
//
#include <hip/hip_runtime.h>
#include <hip/hip_bf16.h>

typedef __attribute__((ext_vector_type(16))) __bf16 v16bf;
typedef __attribute__((ext_vector_type(8)))  float  v8f;
typedef __attribute__((ext_vector_type(4)))  __bf16 v4bf;

#define N_ROWS   65536          // B*T
#define KDIM     512            // FDIM
#define NCOLS    640            // G*V
#define VDIM     320            // V
#define DDIM     128            // VAR_DIM
#define ROWS_WG  16
#define Q_ELEMS  (65536ull * 256ull)

// ---------------- prep: W fp32 -> bf16, zero global softmax accumulator ----
__global__ void vq_prep(const float* __restrict__ W,
                        __bf16* __restrict__ Wb,
                        float* __restrict__ gacc) {
    int i = blockIdx.x * 256 + threadIdx.x;
    if (i < NCOLS * KDIM) Wb[i] = (__bf16)W[i];
    if (blockIdx.x == 0) {
        gacc[threadIdx.x] = 0.0f;
        if (threadIdx.x < VDIM - 256) gacc[256 + threadIdx.x] = 0.0f;
    }
}

// ---------------- main: bf16 WMMA GEMM + softmax/ppl accum + argmax gather -
__global__ __launch_bounds__(256)
void vq_main(const float* __restrict__ x,
             const float* __restrict__ gumbel,
             const __bf16* __restrict__ Wb,
             const float* __restrict__ bias,
             const float* __restrict__ cb,
             float* __restrict__ gacc,
             float* __restrict__ outQ) {
    __shared__ __align__(32) __bf16 lsA[ROWS_WG * KDIM];       // 16 KB
    __shared__ __align__(16) float  lsLogits[ROWS_WG * NCOLS]; // 40 KB
    __shared__ float lsAcc[VDIM];
    __shared__ int   lsAmax[ROWS_WG * 2];

    const int tid     = threadIdx.x;
    const int lane    = tid & 31;
    const int wave    = tid >> 5;
    const int rowBase = blockIdx.x * ROWS_WG;

    // Warm L2 with this workgroup's gumbel slab (16 rows * 2 groups * 320 f32)
    {
        const char* gbase = (const char*)(gumbel + (size_t)rowBase * 2 * VDIM);
        __builtin_prefetch(gbase + (size_t)tid * 128, 0, 3);
        if (tid < 64) __builtin_prefetch(gbase + 32768 + (size_t)tid * 128, 0, 3);
    }

    // Zero LDS softmax accumulator
    lsAcc[tid < VDIM ? tid : 0] = 0.0f;           // tid 0..255
    if (tid < VDIM - 256) lsAcc[256 + tid] = 0.0f;

    // Stage A tile: 16 rows x 512 K, fp32 -> bf16 into LDS
#pragma unroll
    for (int k8 = 0; k8 < 8; ++k8) {
        int t   = tid + k8 * 256;   // 0..2047 float4 chunks
        int row = t >> 7;
        int pos = t & 127;
        float4 v = *(const float4*)(x + (size_t)(rowBase + row) * KDIM + pos * 4);
        v4bf h;
        h.x = (__bf16)v.x; h.y = (__bf16)v.y; h.z = (__bf16)v.z; h.w = (__bf16)v.w;
        *(v4bf*)(&lsA[row * KDIM + pos * 4]) = h;
    }
    __syncthreads();

    // ---- GEMM: each wave computes 16 rows x 80 cols (5 WMMA tiles) ----
    const int colBase = wave * 80;
    const int mrow    = lane & 15;
    const int khalf   = (lane >> 4) * 16;   // which 16 of the K=32 slice

    v8f accv[5] = {};
    for (int kb = 0; kb < 16; ++kb) {
        v16bf a = *reinterpret_cast<const v16bf*>(&lsA[mrow * KDIM + kb * 32 + khalf]);
#pragma unroll
        for (int c = 0; c < 5; ++c) {
            int col = colBase + c * 16 + mrow;
            v16bf bm = *reinterpret_cast<const v16bf*>(Wb + (size_t)col * KDIM + kb * 32 + khalf);
            accv[c] = __builtin_amdgcn_wmma_f32_16x16x32_bf16(
                false, a, false, bm, (short)0, accv[c], false, false);
        }
    }

    // Spill logits tile to LDS with bias (C layout: VGPR r -> M=r / r+8)
    const int rhi = (lane >> 4) * 8;
#pragma unroll
    for (int c = 0; c < 5; ++c) {
        int col  = colBase + c * 16 + mrow;
        float bb = bias[col];
#pragma unroll
        for (int r = 0; r < 8; ++r)
            lsLogits[(r + rhi) * NCOLS + col] = accv[c][r] + bb;
    }
    __syncthreads();

    // ---- per (row,group): softmax accum + gumbel argmax; 32 tasks, 4/wave --
#pragma unroll
    for (int j = 0; j < 4; ++j) {
        int t   = wave * 4 + j;
        int row = t >> 1;
        int g   = t & 1;
        const float* lrow = &lsLogits[row * NCOLS + g * VDIM];

        float vals[10];
        float m = -3.4e38f;
#pragma unroll
        for (int i = 0; i < 10; ++i) { vals[i] = lrow[lane + 32 * i]; m = fmaxf(m, vals[i]); }
#pragma unroll
        for (int off = 16; off; off >>= 1) m = fmaxf(m, __shfl_xor(m, off));

        float e[10];
        float s = 0.0f;
#pragma unroll
        for (int i = 0; i < 10; ++i) { e[i] = __expf(vals[i] - m); s += e[i]; }
#pragma unroll
        for (int off = 16; off; off >>= 1) s += __shfl_xor(s, off);
        float inv = 1.0f / s;
#pragma unroll
        for (int i = 0; i < 10; ++i) atomicAdd(&lsAcc[lane + 32 * i], e[i] * inv);

        // argmax(logits + gumbel)  (TAU/softmax are monotone -> same argmax)
        const float* grow = gumbel + ((size_t)(rowBase + row) * 2 + g) * VDIM;
        float mv = -3.4e38f; int mi = 0;
#pragma unroll
        for (int i = 0; i < 10; ++i) {
            int v = lane + 32 * i;
            float tv = vals[i] + grow[v];
            if (tv > mv) { mv = tv; mi = v; }
        }
#pragma unroll
        for (int off = 16; off; off >>= 1) {
            float ov = __shfl_xor(mv, off);
            int   oi = __shfl_xor(mi, off);
            if (ov > mv || (ov == mv && oi < mi)) { mv = ov; mi = oi; }
        }
        if (lane == 0) lsAmax[row * 2 + g] = mi;
    }
    __syncthreads();

    // Flush workgroup softmax sums to global accumulator (L2 f32 atomics)
    __hip_atomic_fetch_add(&gacc[tid], lsAcc[tid],
                           __ATOMIC_RELAXED, __HIP_MEMORY_SCOPE_AGENT);
    if (tid < VDIM - 256)
        __hip_atomic_fetch_add(&gacc[256 + tid], lsAcc[256 + tid],
                               __ATOMIC_RELAXED, __HIP_MEMORY_SCOPE_AGENT);

    // Gather codebook rows: 16 rows * 2 groups * 32 float4 = 1024 chunks
#pragma unroll
    for (int j = 0; j < 4; ++j) {
        int t   = tid + j * 256;
        int row = t >> 6;
        int g   = (t >> 5) & 1;
        int ch  = t & 31;
        int idx = lsAmax[row * 2 + g];
        float4 v = *((const float4*)(cb + ((size_t)(g * VDIM + idx)) * DDIM) + ch);
        *((float4*)(outQ + ((size_t)(rowBase + row)) * 256 + g * DDIM) + ch) = v;
    }
}

// ---------------- finalize: perplexity scalar --------------------------------
__global__ void vq_finalize(const float* __restrict__ gacc, float* __restrict__ outP) {
    __shared__ float part[10];
    int tid   = threadIdx.x;  // 320 threads
    float avg = gacc[tid] * (1.0f / 131072.0f);
    float term = avg * __logf(avg + 1e-7f);
#pragma unroll
    for (int off = 16; off; off >>= 1) term += __shfl_xor(term, off);
    if ((tid & 31) == 0) part[tid >> 5] = term;
    __syncthreads();
    if (tid == 0) {
        float s = 0.0f;
#pragma unroll
        for (int i = 0; i < 10; ++i) s += part[i];
        *outP = __expf(-s);
    }
}

extern "C" void kernel_launch(void* const* d_in, const int* in_sizes, int n_in,
                              void* d_out, int out_size, void* d_ws, size_t ws_size,
                              hipStream_t stream) {
    const float* x   = (const float*)d_in[0];
    const float* gum = (const float*)d_in[1];
    const float* W   = (const float*)d_in[2];
    const float* b   = (const float*)d_in[3];
    const float* cb  = (const float*)d_in[4];
    float* out = (float*)d_out;

    __bf16* Wb  = (__bf16*)d_ws;
    float* gacc = (float*)((char*)d_ws + (size_t)NCOLS * KDIM * 2);

    vq_prep<<<(NCOLS * KDIM + 255) / 256, 256, 0, stream>>>(W, Wb, gacc);
    vq_main<<<N_ROWS / ROWS_WG, 256, 0, stream>>>(x, gum, Wb, b, cb, gacc, out);
    vq_finalize<<<1, VDIM, 0, stream>>>(gacc, out + Q_ELEMS);
}
